// CausalAttn_32031866093894
// MI455X (gfx1250) — matrix operand order repaired and checked by reading, not compiled, not run on hardware
//
#include <hip/hip_runtime.h>
#include <hip/hip_bf16.h>
#include <stdint.h>

// ---- CDNA5 WMMA types -------------------------------------------------------
typedef __bf16 bf16;
typedef __attribute__((ext_vector_type(16))) __bf16 v16bf;
typedef __attribute__((ext_vector_type(8)))  float  v8f;

union Frag16 { v16bf v; uint32_t u[8]; };

__device__ __forceinline__ v8f wmma_bf16(v16bf a, v16bf b, v8f c) {
    // D = A(16x32) * B(32x16) + C, f32 accumulate
    return __builtin_amdgcn_wmma_f32_16x16x32_bf16(
        /*neg_a=*/false, a, /*neg_b=*/false, b,
        /*c_mod=*/(short)0, c, /*reuse_a=*/false, /*reuse_b=*/false);
}

__device__ __forceinline__ bf16 f32_to_bf16(float f) {
    union { float f; uint32_t u; } a; a.f = f;
    uint32_t r = a.u + 0x7FFFu + ((a.u >> 16) & 1u);   // round-to-nearest-even
    union { unsigned short s; bf16 b; } o; o.s = (unsigned short)(r >> 16);
    return o.b;
}

// ---- CDNA5 async global->LDS copy (ASYNCcnt-tracked, no VGPR staging) -------
// LDS aperture keeps the wave-relative LDS byte offset in addr[31:0], so a
// truncated flat pointer is the VDST operand the instruction expects.
__device__ __forceinline__ uint32_t lds_off(const void* p) {
    return (uint32_t)(size_t)p;
}
__device__ __forceinline__ void async_copy_b128(uint32_t dst_lds, const void* src) {
    asm volatile("global_load_async_to_lds_b128 %0, %1, off"
                 :: "v"(dst_lds), "v"(src) : "memory");
}
__device__ __forceinline__ void wait_async_le4() {
    asm volatile("s_wait_asynccnt 0x4" ::: "memory");
}
__device__ __forceinline__ void wait_async_le0() {
    asm volatile("s_wait_asynccnt 0x0" ::: "memory");
}

// A fragment: matrix stored row-major [16 M][ld K-contig], covers K=0..31 at base.
// Lane layout (ISA 7.12.2): m = lane&15; elems 0..7 -> K = half*8 + e;
// elems 8..15 -> K = 16 + half*8 + (e-8).
__device__ __forceinline__ v16bf load_A_frag(const bf16* base, int ld) {
    int lane = threadIdx.x & 31;
    int m = lane & 15, h = lane >> 4;
    Frag16 f;
    const uint4* p0 = (const uint4*)(base + (size_t)m * ld + h * 8);
    const uint4* p1 = (const uint4*)(base + (size_t)m * ld + 16 + h * 8);
    uint4 t0 = *p0, t1 = *p1;
    f.u[0] = t0.x; f.u[1] = t0.y; f.u[2] = t0.z; f.u[3] = t0.w;
    f.u[4] = t1.x; f.u[5] = t1.y; f.u[6] = t1.z; f.u[7] = t1.w;
    return f.v;
}

// B fragment: B^T stored row-major [16 N][ld K-contig], covers K=0..31 at base.
// Lane layout: n = lane&15; elems 0..7 -> K = half*8 + e;
// elems 8..15 -> K = 16 + half*8 + (e-8).
__device__ __forceinline__ v16bf load_B_frag(const bf16* base, int ld) {
    int lane = threadIdx.x & 31;
    int n = lane & 15, h = lane >> 4;
    Frag16 f;
    const uint4* p0 = (const uint4*)(base + (size_t)n * ld + h * 8);
    const uint4* p1 = (const uint4*)(base + (size_t)n * ld + 16 + h * 8);
    uint4 t0 = *p0, t1 = *p1;
    f.u[0] = t0.x; f.u[1] = t0.y; f.u[2] = t0.z; f.u[3] = t0.w;
    f.u[4] = t1.x; f.u[5] = t1.y; f.u[6] = t1.z; f.u[7] = t1.w;
    return f.v;
}

// ---- problem constants ------------------------------------------------------
#define BB 2
#define TT 2048
#define CC 1024
#define HH 16
#define HD 64

// ---- elementwise f32 -> bf16 ------------------------------------------------
__global__ __launch_bounds__(256) void convert_bf16(const float* __restrict__ in,
                                                    bf16* __restrict__ out, int n) {
    int i = (blockIdx.x * 256 + threadIdx.x) * 4;
    if (i + 3 < n) {
        float4 v = *(const float4*)(in + i);
        union { bf16 b[4]; uint2 u; } o;
        o.b[0] = f32_to_bf16(v.x); o.b[1] = f32_to_bf16(v.y);
        o.b[2] = f32_to_bf16(v.z); o.b[3] = f32_to_bf16(v.w);
        *(uint2*)(out + i) = o.u;
    }
}

// ---- f32 [R][Cc] -> bf16 [Cc][R] (transpose for B^T fragment layout) -------
__global__ __launch_bounds__(256) void transpose_to_bf16(const float* __restrict__ in,
                                                         bf16* __restrict__ out,
                                                         int R, int Cc) {
    __shared__ float tile[32][33];
    int bx = blockIdx.x * 32;   // column base in input
    int by = blockIdx.y * 32;   // row base in input
    int tx = threadIdx.x & 31, ty = threadIdx.x >> 5;  // 32 x 8
#pragma unroll
    for (int i = 0; i < 32; i += 8)
        tile[ty + i][tx] = in[(size_t)(by + ty + i) * Cc + bx + tx];
    __syncthreads();
#pragma unroll
    for (int i = 0; i < 32; i += 8)
        out[(size_t)(bx + ty + i) * R + by + tx] = f32_to_bf16(tile[tx][ty + i]);
}

// ---- tiled bf16 WMMA GEMM: Out = A[M][K] * BT[N][K]^T + bias ----------------
// 128x128 block tile, 8 waves as 2(M) x 4(N); each wave: 4 M-tiles x 2 N-tiles.
// Double-buffered LDS filled by async global->LDS DMA (ASYNCcnt).
// mode 0: store f32 to outF [M][N]
// mode 1: qkv epilogue -> q[B,H,T,HD], k[B,H,T,HD], vT[B,H,HD,T] (bf16)
__global__ __launch_bounds__(256)
void gemm_bf16(const bf16* __restrict__ A, const bf16* __restrict__ BT,
               const float* __restrict__ bias, int M, int N, int K, int mode,
               float* __restrict__ outF,
               bf16* __restrict__ qb, bf16* __restrict__ kb, bf16* __restrict__ vt) {
    __shared__ __align__(16) bf16 sA[2][128 * 32];   // [m][k]
    __shared__ __align__(16) bf16 sB[2][128 * 32];   // [n][k]
    const int tid = threadIdx.x;
    const int wave = tid >> 5, lane = tid & 31;
    const int lane16 = lane & 15, lh = lane >> 4;
    const int wm = wave >> 2;      // 0..1 : 64-row half
    const int wn = wave & 3;       // 0..3 : 32-col quarter
    const int m0 = blockIdx.y * 128;
    const int n0 = blockIdx.x * 128;

    // Each thread copies 4x16B per stage (2 for A, 2 for B).
    auto issue_stage = [&](int k0, int buf) {
#pragma unroll
        for (int c = 0; c < 2; ++c) {
            int ch = c * 256 + tid;          // 0..511
            int r = ch >> 2, cs = (ch & 3) * 8;
            async_copy_b128(lds_off(&sA[buf][r * 32 + cs]),
                            &A[(size_t)(m0 + r) * K + k0 + cs]);
            async_copy_b128(lds_off(&sB[buf][r * 32 + cs]),
                            &BT[(size_t)(n0 + r) * K + k0 + cs]);
        }
    };

    v8f acc[4][2];
    const v8f vzero = {0.f, 0.f, 0.f, 0.f, 0.f, 0.f, 0.f, 0.f};
#pragma unroll
    for (int i = 0; i < 4; ++i)
#pragma unroll
        for (int j = 0; j < 2; ++j) acc[i][j] = vzero;

    const int nIter = K >> 5;
    issue_stage(0, 0);
    for (int i = 0; i < nIter; ++i) {
        if (i + 1 < nIter) {
            issue_stage((i + 1) << 5, (i + 1) & 1);  // prefetch next tile (DMA overlaps WMMA)
            wait_async_le4();                        // stage i done (in-order completion)
        } else {
            wait_async_le0();
        }
        __syncthreads();
        const bf16* a0 = &sA[i & 1][(wm * 64) * 32];
        const bf16* b0 = &sB[i & 1][(wn * 32) * 32];
        v16bf aF[4], bF[2];
#pragma unroll
        for (int mt = 0; mt < 4; ++mt) aF[mt] = load_A_frag(a0 + mt * 16 * 32, 32);
#pragma unroll
        for (int nt = 0; nt < 2; ++nt) bF[nt] = load_B_frag(b0 + nt * 16 * 32, 32);
#pragma unroll
        for (int mt = 0; mt < 4; ++mt)
#pragma unroll
            for (int nt = 0; nt < 2; ++nt)
                acc[mt][nt] = wmma_bf16(aF[mt], bF[nt], acc[mt][nt]);
        __syncthreads();
    }

#pragma unroll
    for (int nt = 0; nt < 2; ++nt) {
        const int nw = n0 + wn * 32 + nt * 16 + lane16;
        const float bv = bias[nw];
        if (mode == 0) {
#pragma unroll
            for (int mt = 0; mt < 4; ++mt)
#pragma unroll
                for (int r = 0; r < 8; ++r) {
                    int m = m0 + wm * 64 + mt * 16 + r + 8 * lh;
                    outF[(size_t)m * N + nw] = acc[mt][nt][r] + bv;
                }
        } else {
            // n in [0,3C): section 0=k, 1=q, 2=v (reference split order)
            int sect = nw / CC, cc = nw - sect * CC;
            int hh = cc >> 6, d = cc & 63;
#pragma unroll
            for (int mt = 0; mt < 4; ++mt)
#pragma unroll
                for (int r = 0; r < 8; ++r) {
                    int m = m0 + wm * 64 + mt * 16 + r + 8 * lh;
                    int b = m >> 11, t = m & (TT - 1);
                    bf16 v = f32_to_bf16(acc[mt][nt][r] + bv);
                    size_t bh = (size_t)(b * HH + hh);
                    if (sect == 0)      kb[(bh * TT + t) * HD + d] = v;
                    else if (sect == 1) qb[(bh * TT + t) * HD + d] = v;
                    else                vt[(bh * HD + d) * TT + t] = v;
                }
        }
    }
}

// ---- causal flash attention: per (b,h) x 128 q rows -------------------------
__global__ __launch_bounds__(256)
void flash_attn(const bf16* __restrict__ qb, const bf16* __restrict__ kb,
                const bf16* __restrict__ vt, bf16* __restrict__ ao) {
    __shared__ __align__(16) bf16 sK[64 * 64];      // [k_local][d]
    __shared__ __align__(16) bf16 sV[64 * 64];      // [d][k_local]  (V^T)
    __shared__ __align__(16) bf16 sP[8][16 * 32];   // per-wave P tile [m][32]
    const int tid = threadIdx.x;
    const int wave = tid >> 5, lane = tid & 31;
    const int lane16 = lane & 15, lh = lane >> 4;
    const int bh = blockIdx.y;                 // b*H + h
    const int qt0 = blockIdx.x * 128;
    const int qr0 = qt0 + wave * 16;

    const bf16* qbase = qb + ((size_t)bh * TT + qr0) * HD;
    v16bf qf0 = load_A_frag(qbase, HD);        // K(d) = 0..31
    v16bf qf1 = load_A_frag(qbase + 32, HD);   // K(d) = 32..63

    const v8f vzero = {0.f, 0.f, 0.f, 0.f, 0.f, 0.f, 0.f, 0.f};
    v8f o[4];
#pragma unroll
    for (int i = 0; i < 4; ++i) o[i] = vzero;
    float mrow[8], lrow[8];
#pragma unroll
    for (int r = 0; r < 8; ++r) { mrow[r] = -__builtin_inff(); lrow[r] = 0.f; }

    const int kend = qt0 + 128;
    for (int j0 = 0; j0 < kend; j0 += 64) {
        // async DMA the 64-wide K and V^T chunks into LDS (4x16B per thread)
#pragma unroll
        for (int c = 0; c < 2; ++c) {
            int ch = c * 256 + tid;          // 64 rows * 8 chunks
            int r = ch >> 3, cs = (ch & 7) * 8;
            async_copy_b128(lds_off(&sK[r * 64 + cs]),
                            &kb[((size_t)bh * TT + j0 + r) * HD + cs]);
            async_copy_b128(lds_off(&sV[r * 64 + cs]),
                            &vt[((size_t)bh * HD + r) * TT + j0 + cs]);
        }
        wait_async_le0();
        __syncthreads();

#pragma unroll
        for (int sub = 0; sub < 2; ++sub) {   // 32 k-columns per pass
            v8f s[2];
#pragma unroll
            for (int tt = 0; tt < 2; ++tt) {  // two 16-col S tiles
                const bf16* kb0 = &sK[(sub * 32 + tt * 16) * 64];
                v8f st = vzero;
                st = wmma_bf16(qf0, load_B_frag(kb0, 64), st);
                st = wmma_bf16(qf1, load_B_frag(kb0 + 32, 64), st);
                s[tt] = st;
            }
            const int kc0 = j0 + sub * 32;
            float alpha[8];
#pragma unroll
            for (int r = 0; r < 8; ++r) {
                int qg = qr0 + r + 8 * lh;
                float s0 = s[0][r] * 0.125f;   // HD^-0.5
                float s1 = s[1][r] * 0.125f;
                s0 = (kc0 + lane16 <= qg)      ? s0 : -__builtin_inff();
                s1 = (kc0 + 16 + lane16 <= qg) ? s1 : -__builtin_inff();
                float rm = fmaxf(s0, s1);
                rm = fmaxf(rm, __shfl_xor(rm, 1));
                rm = fmaxf(rm, __shfl_xor(rm, 2));
                rm = fmaxf(rm, __shfl_xor(rm, 4));
                rm = fmaxf(rm, __shfl_xor(rm, 8));
                float mn = fmaxf(mrow[r], rm);
                alpha[r] = __expf(mrow[r] - mn);
                float p0 = __expf(s0 - mn);
                float p1 = __expf(s1 - mn);
                float rs = p0 + p1;
                rs += __shfl_xor(rs, 1);
                rs += __shfl_xor(rs, 2);
                rs += __shfl_xor(rs, 4);
                rs += __shfl_xor(rs, 8);
                lrow[r] = lrow[r] * alpha[r] + rs;
                mrow[r] = mn;
                bf16* pp = &sP[wave][(r + 8 * lh) * 32];
                pp[lane16]      = f32_to_bf16(p0);
                pp[16 + lane16] = f32_to_bf16(p1);
            }
#pragma unroll
            for (int dt = 0; dt < 4; ++dt)
#pragma unroll
                for (int r = 0; r < 8; ++r) o[dt][r] *= alpha[r];
            // Intra-wave visibility of sP writes: drain this wave's DS ops
            // using the CDNA5 split DS counter (no workgroup barrier needed).
            asm volatile("s_wait_dscnt 0" ::: "memory");
            v16bf pf = load_A_frag(&sP[wave][0], 32);
#pragma unroll
            for (int dt = 0; dt < 4; ++dt) {
                v16bf vf = load_B_frag(&sV[dt * 16 * 64 + sub * 32], 64);
                o[dt] = wmma_bf16(pf, vf, o[dt]);
            }
        }
        __syncthreads();
    }

    const int b = bh >> 4, hh = bh & 15;
#pragma unroll
    for (int r = 0; r < 8; ++r) {
        int qg = qr0 + r + 8 * lh;
        float inv = 1.0f / lrow[r];
        size_t base = ((size_t)b * TT + qg) * CC + hh * HD;
#pragma unroll
        for (int dt = 0; dt < 4; ++dt)
            ao[base + dt * 16 + lane16] = f32_to_bf16(o[dt][r] * inv);
    }
}

// ---- host side --------------------------------------------------------------
extern "C" void kernel_launch(void* const* d_in, const int* in_sizes, int n_in,
                              void* d_out, int out_size, void* d_ws, size_t ws_size,
                              hipStream_t stream) {
    const float* x      = (const float*)d_in[0];   // [B,T,C]
    const float* attn_w = (const float*)d_in[1];   // [C,3C]
    const float* attn_b = (const float*)d_in[2];   // [3C]
    const float* proj_w = (const float*)d_in[3];   // [C,C]
    const float* proj_b = (const float*)d_in[4];   // [C]
    float* out = (float*)d_out;                    // [B,T,C] f32

    char* ws = (char*)d_ws;
    const size_t SZ_XB = (size_t)BB * TT * CC * 2;        // 8 MB
    const size_t SZ_WT = (size_t)3 * CC * CC * 2;         // 6 MB
    const size_t SZ_PT = (size_t)CC * CC * 2;             // 2 MB
    const size_t SZ_H  = (size_t)BB * HH * TT * HD * 2;   // 8 MB each
    bf16* xb = (bf16*)(ws);
    bf16* wT = (bf16*)(ws + SZ_XB);
    bf16* pT = (bf16*)(ws + SZ_XB + SZ_WT);
    bf16* qb = (bf16*)(ws + SZ_XB + SZ_WT + SZ_PT);
    bf16* kb = (bf16*)(ws + SZ_XB + SZ_WT + SZ_PT + SZ_H);
    bf16* vt = (bf16*)(ws + SZ_XB + SZ_WT + SZ_PT + 2 * SZ_H);
    bf16* ao = (bf16*)(ws + SZ_XB + SZ_WT + SZ_PT + 3 * SZ_H);
    (void)in_sizes; (void)n_in; (void)out_size; (void)ws_size;

    const int M = BB * TT;   // 4096

    convert_bf16<<<(M * CC) / (256 * 4), 256, 0, stream>>>(x, xb, M * CC);
    transpose_to_bf16<<<dim3(3 * CC / 32, CC / 32), 256, 0, stream>>>(attn_w, wT, CC, 3 * CC);
    transpose_to_bf16<<<dim3(CC / 32, CC / 32), 256, 0, stream>>>(proj_w, pT, CC, CC);

    // qkv = x @ attn_w + attn_b, scattered to q/k/v^T head layouts
    gemm_bf16<<<dim3(3 * CC / 128, M / 128), 256, 0, stream>>>(
        xb, wT, attn_b, M, 3 * CC, CC, 1, nullptr, qb, kb, vt);

    // causal flash attention -> ao [B,T,C] bf16
    flash_attn<<<dim3(TT / 128, BB * HH), 256, 0, stream>>>(qb, kb, vt, ao);

    // out = ao @ proj_w + proj_b (f32)
    gemm_bf16<<<dim3(CC / 128, M / 128), 256, 0, stream>>>(
        ao, pT, proj_b, M, CC, CC, 0, out, nullptr, nullptr, nullptr);
}
